// SMEEG_34797825032958
// MI455X (gfx1250) — matrix operand order repaired and checked by reading, not compiled, not run on hardware
//
#include <hip/hip_runtime.h>
#include <math.h>

typedef __attribute__((ext_vector_type(16))) _Float16 v16h;
typedef __attribute__((ext_vector_type(8)))  float    v8f;

#define LAYERS 4
#define EMB   256
#define TOKN  8
#define GC    32
#define DI    64
#define DSS   16
#define DTR   2
#define BB    16
#define LL    512
#define NSEQ  (BB * LL)          // 8192 sequences per layer pass

__device__ __forceinline__ float siluf(float v)     { return v / (1.0f + __expf(-v)); }
__device__ __forceinline__ float softplusf(float v) { return (v > 20.0f) ? v : log1pf(__expf(v)); }

// A-fragment K map (16-bit A 16x32, ISA 7.12.2): halves 0-7 -> K 0-7 (lo lanes)/8-15 (hi),
// halves 8-15 -> K 16-23 (lo)/24-31 (hi)
__device__ __forceinline__ int amapK(int j, int lhi) {
  return j + ((j < 8) ? (lhi ? 8 : 0) : (lhi ? 16 : 8));
}
// B-fragment K map (16-bit B 32x16): lane%16 = N, same K order as amapK
__device__ __forceinline__ int bmapK(int j, int lhi) { return amapK(j, lhi); }

// ---------------------------------------------------------------------------
// Fused Mamba block: one wave handles a 16-token tile = 2 sequences.
// h (in) and xr (out) are [NSEQ][TOKN][GC] fp32 (== [B,L,C] contiguous).
// ---------------------------------------------------------------------------
__global__ void __launch_bounds__(256) mamba_kernel(
    const float* __restrict__ h,     // [NSEQ,8,32]
    const float* __restrict__ Wi,    // [128,32]
    const float* __restrict__ Wc,    // [64,4]
    const float* __restrict__ bc,    // [64]
    const float* __restrict__ Wx,    // [34,64]
    const float* __restrict__ Wdt,   // [64,2]
    const float* __restrict__ bdt,   // [64]
    const float* __restrict__ Alog,  // [64,16]
    const float* __restrict__ Dp,    // [64]
    const float* __restrict__ Wo,    // [32,64]
    float* __restrict__ xr)          // [NSEQ,8,32]
{
  __shared__ _Float16 lds_x[8][16][DI];   // per-wave activation staging (A operand)
  __shared__ float    lds_B[8][16][DSS];  // Bm per row(=token-in-tile)
  __shared__ float    lds_C[8][16][DSS];  // Cm per row

  const int lane = threadIdx.x & 31;
  const int wid  = threadIdx.x >> 5;
  const int n    = lane & 15;       // column index within 16-wide tiles
  const int lhi  = lane >> 4;       // 0/1 : also sequence-in-tile for D-layout rows
  const int tile = blockIdx.x * 8 + wid;
  const int seq0 = tile * 2;
  const float* ubase = h + (size_t)seq0 * (TOKN * GC);  // contiguous [16][32]

  // ---- A fragment of u : rows = 16 tokens (2 seqs), K = 32 model dims
  v16h a_u;
  #pragma unroll
  for (int j = 0; j < 16; ++j)
    a_u[j] = (_Float16)ubase[n * GC + amapK(j, lhi)];

  // ---- in_proj: xz[16,128] = u @ Wi^T   (8 column tiles of 16)
  v8f acc[8];
  #pragma unroll
  for (int nt = 0; nt < 8; ++nt) {
    v16h b;
    const float* wrow = Wi + (size_t)(16 * nt + n) * GC;   // B[k,n] = Wi[col,k]
    #pragma unroll
    for (int j = 0; j < 16; ++j) b[j] = (_Float16)wrow[bmapK(j, lhi)];
    v8f cz = {};
    acc[nt] = __builtin_amdgcn_wmma_f32_16x16x32_f16(
        false, a_u, false, b, (short)0, cz, false, false);
  }

  // D-layout: lane holds channel c = 16*nt + n; acc[nt][t] = value at token t of its seq.
  // ---- causal depthwise conv (k=4) + bias + SiLU, fully in-lane
  float xact[4][8];
  #pragma unroll
  for (int nt = 0; nt < 4; ++nt) {
    const int c = 16 * nt + n;
    const float w0 = Wc[c*4+0], w1 = Wc[c*4+1], w2 = Wc[c*4+2], w3 = Wc[c*4+3];
    const float bias = bc[c];
    #pragma unroll
    for (int t = 0; t < 8; ++t) {
      float s = acc[nt][t] * w3;
      if (t >= 1) s += acc[nt][t-1] * w2;
      if (t >= 2) s += acc[nt][t-2] * w1;
      if (t >= 3) s += acc[nt][t-3] * w0;
      xact[nt][t] = siluf(s + bias);
    }
  }

  // stage activated x into LDS [16 rows][64 ch] as f16 for the next A operand
  #pragma unroll
  for (int nt = 0; nt < 4; ++nt) {
    const int c = 16 * nt + n;
    #pragma unroll
    for (int t = 0; t < 8; ++t)
      lds_x[wid][lhi * 8 + t][c] = (_Float16)xact[nt][t];
  }
  __syncthreads();

  // ---- x_proj: x_dbl[16,34] = x @ Wx^T  (3 col tiles x 2 K-chunks)
  v8f accd[3];
  #pragma unroll
  for (int ct = 0; ct < 3; ++ct) {
    v8f cz = {};
    #pragma unroll
    for (int kc = 0; kc < 2; ++kc) {
      v16h a;
      #pragma unroll
      for (int j = 0; j < 16; ++j)
        a[j] = lds_x[wid][n][32 * kc + amapK(j, lhi)];
      v16h b;
      const int g = 16 * ct + n;
      #pragma unroll
      for (int j = 0; j < 16; ++j)
        b[j] = (g < 34) ? (_Float16)Wx[(size_t)g * DI + 32 * kc + bmapK(j, lhi)]
                        : (_Float16)0.0f;
      cz = __builtin_amdgcn_wmma_f32_16x16x32_f16(
          false, a, false, b, (short)0, cz, false, false);
    }
    accd[ct] = cz;
  }

  // scatter Bm (cols 2..17) and Cm (cols 18..33) into LDS, per row
  #pragma unroll
  for (int v = 0; v < 8; ++v) {
    const int row = v + lhi * 8;
    if (n >= 2) lds_B[wid][row][n - 2]  = accd[0][v];
    if (n < 2)  lds_B[wid][row][14 + n] = accd[1][v];
    else        lds_C[wid][row][n - 2]  = accd[1][v];
    if (n < 2)  lds_C[wid][row][14 + n] = accd[2][v];
  }
  __syncthreads();

  // dt (cols 0,1) broadcast: lane 16*s+{0,1} holds seq s tokens in accd[0][t]
  float dt0[8], dt1[8];
  #pragma unroll
  for (int t = 0; t < 8; ++t) {
    dt0[t] = __shfl(accd[0][t], lhi * 16 + 0, 32);
    dt1[t] = __shfl(accd[0][t], lhi * 16 + 1, 32);
  }

  // ---- selective scan: per lane-channel, 16-state recurrence over T=8, then z-gate
  float yout[4][8];
  #pragma unroll
  for (int nt = 0; nt < 4; ++nt) {
    const int c = 16 * nt + n;
    const float wdt0 = Wdt[c * 2 + 0], wdt1 = Wdt[c * 2 + 1], bd = bdt[c];
    const float Dc = Dp[c];
    float Aa[DSS];
    #pragma unroll
    for (int si = 0; si < DSS; ++si) Aa[si] = -__expf(Alog[c * DSS + si]);
    float hs[DSS];
    #pragma unroll
    for (int si = 0; si < DSS; ++si) hs[si] = 0.0f;
    #pragma unroll
    for (int t = 0; t < 8; ++t) {
      const int row = lhi * 8 + t;
      const float dl = softplusf(dt0[t] * wdt0 + dt1[t] * wdt1 + bd);
      const float xv = xact[nt][t];
      const float dx = dl * xv;
      float y = 0.0f;
      #pragma unroll
      for (int si = 0; si < DSS; ++si) {
        const float dA = __expf(dl * Aa[si]);
        hs[si] = dA * hs[si] + dx * lds_B[wid][row][si];
        y += hs[si] * lds_C[wid][row][si];
      }
      y += xv * Dc;
      yout[nt][t] = y * siluf(acc[nt + 4][t]);   // z tiles 4..7 share lane<->channel map
    }
  }
  __syncthreads();

  // stage gated y into LDS (reuse lds_x) for out_proj A operand
  #pragma unroll
  for (int nt = 0; nt < 4; ++nt) {
    const int c = 16 * nt + n;
    #pragma unroll
    for (int t = 0; t < 8; ++t)
      lds_x[wid][lhi * 8 + t][c] = (_Float16)yout[nt][t];
  }
  __syncthreads();

  // ---- out_proj: out[16,32] = y @ Wo^T  (2 col tiles x 2 K-chunks), store to xr
  #pragma unroll
  for (int ct = 0; ct < 2; ++ct) {
    v8f cz = {};
    #pragma unroll
    for (int kc = 0; kc < 2; ++kc) {
      v16h a;
      #pragma unroll
      for (int j = 0; j < 16; ++j)
        a[j] = lds_x[wid][n][32 * kc + amapK(j, lhi)];
      v16h b;
      const float* worow = Wo + (size_t)(16 * ct + n) * DI;
      #pragma unroll
      for (int j = 0; j < 16; ++j)
        b[j] = (_Float16)worow[32 * kc + bmapK(j, lhi)];
      cz = __builtin_amdgcn_wmma_f32_16x16x32_f16(
          false, a, false, b, (short)0, cz, false, false);
    }
    #pragma unroll
    for (int v = 0; v < 8; ++v) {
      // D-layout: lane col = 16*ct+n, row v + 8*lhi -> seq lhi, token v
      xr[(size_t)(seq0 + lhi) * (TOKN * GC) + v * GC + 16 * ct + n] = cz[v];
    }
  }
}

// ---------------------------------------------------------------------------
// GroupNorm stats: one block per (batch, group); reduce over 512*64 elements.
// ---------------------------------------------------------------------------
__global__ void __launch_bounds__(256) gn_stats_kernel(const float* __restrict__ xr,
                                                       float* __restrict__ stats)
{
  const int b = blockIdx.x >> 2;
  const int g = blockIdx.x & 3;
  const float* base = xr + (size_t)b * LL * EMB + g * 64;
  float s = 0.f, s2 = 0.f;
  for (int i = threadIdx.x; i < LL * 64; i += 256) {
    const int l = i >> 6, c = i & 63;
    const float v = base[(size_t)l * EMB + c];
    s += v; s2 += v * v;
  }
  __shared__ float sh[256], sh2[256];
  sh[threadIdx.x] = s; sh2[threadIdx.x] = s2;
  __syncthreads();
  for (int off = 128; off > 0; off >>= 1) {
    if (threadIdx.x < off) {
      sh[threadIdx.x]  += sh[threadIdx.x + off];
      sh2[threadIdx.x] += sh2[threadIdx.x + off];
    }
    __syncthreads();
  }
  if (threadIdx.x == 0) {
    const float inv = 1.0f / (LL * 64.0f);
    const float mu  = sh[0] * inv;
    const float var = sh2[0] * inv - mu * mu;
    stats[blockIdx.x * 2 + 0] = mu;
    stats[blockIdx.x * 2 + 1] = rsqrtf(var + 1e-5f);
  }
}

// GroupNorm apply + SiLU + residual accumulate into h
__global__ void __launch_bounds__(256) gn_apply_kernel(float* __restrict__ h,
    const float* __restrict__ xr, const float* __restrict__ stats,
    const float* __restrict__ gamma, const float* __restrict__ beta)
{
  const size_t idx = (size_t)blockIdx.x * 256 + threadIdx.x;
  const int c = (int)(idx % EMB);
  const int b = (int)(idx / ((size_t)LL * EMB));
  const int g = c >> 6;
  const float mu = stats[(b * 4 + g) * 2 + 0];
  const float rs = stats[(b * 4 + g) * 2 + 1];
  const float v  = (xr[idx] - mu) * rs * gamma[c] + beta[c];
  h[idx] += siluf(v);
}

__global__ void __launch_bounds__(256) copy_kernel(float* __restrict__ dst,
                                                   const float* __restrict__ src, size_t nelem)
{
  const size_t idx = (size_t)blockIdx.x * 256 + threadIdx.x;
  if (idx < nelem) dst[idx] = src[idx];
}

// ---------------------------------------------------------------------------
extern "C" void kernel_launch(void* const* d_in, const int* in_sizes, int n_in,
                              void* d_out, int out_size, void* d_ws, size_t ws_size,
                              hipStream_t stream)
{
  (void)in_sizes; (void)n_in; (void)out_size; (void)ws_size;
  const float* x    = (const float*)d_in[0];
  const float* Wi   = (const float*)d_in[1];
  const float* Wc   = (const float*)d_in[2];
  const float* bcn  = (const float*)d_in[3];
  const float* Wx   = (const float*)d_in[4];
  const float* Wdt  = (const float*)d_in[5];
  const float* bdt  = (const float*)d_in[6];
  const float* Alog = (const float*)d_in[7];
  const float* Dp   = (const float*)d_in[8];
  const float* Wo   = (const float*)d_in[9];
  const float* gam  = (const float*)d_in[10];
  const float* bet  = (const float*)d_in[11];

  float* h     = (float*)d_out;                     // [B,L,C], also final output layout
  float* xr    = (float*)d_ws;                      // [B,L,C] scratch
  float* stats = (float*)((char*)d_ws + sizeof(float) * (size_t)BB * LL * EMB);

  const size_t NE = (size_t)BB * LL * EMB;          // 2,097,152
  copy_kernel<<<(int)((NE + 255) / 256), 256, 0, stream>>>(h, x, NE);

  for (int layer = 0; layer < LAYERS; ++layer) {
    mamba_kernel<<<NSEQ / 16, 256, 0, stream>>>(
        h,
        Wi   + (size_t)layer * 128 * 32,
        Wc   + (size_t)layer * 64 * 4,
        bcn  + (size_t)layer * 64,
        Wx   + (size_t)layer * 34 * 64,
        Wdt  + (size_t)layer * 64 * 2,
        bdt  + (size_t)layer * 64,
        Alog + (size_t)layer * 64 * 16,
        Dp   + (size_t)layer * 64,
        Wo   + (size_t)layer * 32 * 64,
        xr);
    gn_stats_kernel<<<BB * 4, 256, 0, stream>>>(xr, stats);
    gn_apply_kernel<<<(int)((NE + 255) / 256), 256, 0, stream>>>(
        h, xr, stats, gam + (size_t)layer * EMB, bet + (size_t)layer * EMB);
  }
}
